// W8A8Conv2d_15753940041869
// MI455X (gfx1250) — 8-bit matrix operand order repaired and checked by reading, compiled, not run on hardware
//
#include <hip/hip_runtime.h>
#include <stdint.h>

typedef __attribute__((ext_vector_type(8))) int v8i;
typedef __attribute__((ext_vector_type(4))) int v4i;
typedef __attribute__((ext_vector_type(2))) int v2i;

#define CIN  128
#define COUT 128
#define HH   256
#define WW   256
#define BB   8

#define XQ_BYTES ((size_t)BB * HH * WW * CIN)

__global__ __launch_bounds__(256) void quant_act_kernel(
    const float* __restrict__ x, const float* __restrict__ x_scale,
    uint8_t* __restrict__ xq) {
  __shared__ int lqi[64 * 33];
  unsigned char* lq = (unsigned char*)lqi;
  const int t  = threadIdx.x;
  const int pb = blockIdx.x;
  const int w0 = (pb & 3) << 6;
  const int h  = (pb >> 2) & (HH - 1);
  const int b  = pb >> 10;
  const float inv = 1.0f / x_scale[0];

  #pragma unroll
  for (int cc = 0; cc < 4; ++cc) {
    #pragma unroll
    for (int i = 0; i < 8; ++i) {
      int e  = t + (i << 8);
      int cl = e >> 6;
      int p  = e & 63;
      int c  = cc * 32 + cl;
      float v = x[(((size_t)b * CIN + c) * HH + h) * WW + w0 + p];
      float q = rintf(v * inv);
      q = fminf(127.0f, fmaxf(-128.0f, q));
      int qi = (int)q;
      lq[p * 132 + c] = (unsigned char)(qi & 0xff);
    }
  }
  __syncthreads();
  int* xq32 = (int*)xq;
  const size_t nbase = ((size_t)b * HH + h) * WW + w0;
  #pragma unroll
  for (int i = 0; i < 8; ++i) {
    int e = t + (i << 8);
    int p = e >> 5;
    int d = e & 31;
    xq32[(nbase + p) * 32 + d] = lqi[p * 33 + d];
  }
}

__global__ __launch_bounds__(256) void pack_w_kernel(
    const int* __restrict__ w32, uint8_t* __restrict__ wpk) {
  int idx = blockIdx.x * 256 + threadIdx.x;
  if (idx >= COUT * 9 * 32) return;
  int d   = idx & 31;
  int tap = (idx >> 5) % 9;
  int co  = idx / (9 * 32);
  int kh = tap / 3, kw = tap % 3;
  unsigned int pk = 0;
  #pragma unroll
  for (int j = 0; j < 4; ++j) {
    int cin = d * 4 + j;
    int wv = w32[(((co * CIN) + cin) * 3 + kh) * 3 + kw];
    pk |= ((unsigned int)(wv & 0xff)) << (8 * j);
  }
  ((unsigned int*)wpk)[idx] = pk;
}

__global__ __launch_bounds__(256) void conv_wmma_kernel(
    const uint8_t* __restrict__ xq, const uint8_t* __restrict__ wpk,
    const float* __restrict__ w_scale, const float* __restrict__ bias,
    const float* __restrict__ x_scale, float* __restrict__ out) {
  __shared__ int acts[3 * 66 * 36];
  const int t  = threadIdx.x;
  const int bt = blockIdx.x;
  const int w0 = (bt & 3) << 6;
  const int h  = (bt >> 2) & (HH - 1);
  const int b  = bt >> 10;

  const int* xq32 = (const int*)xq;
  #pragma unroll
  for (int i = 0; i < 25; ++i) {
    int idx = t + (i << 8);
    if (idx < 3 * 66 * 32) {
      int d    = idx & 31;
      int cell = idx >> 5;
      int y    = cell / 66;
      int xcol = cell % 66;
      int gy = h + y - 1;
      int gx = w0 + xcol - 1;
      int* lp = &acts[cell * 36 + d];
      if (gy >= 0 && gy < HH && gx >= 0 && gx < WW) {
        const int* gp = &xq32[(((size_t)b * HH + gy) * WW + gx) * 32 + d];
        unsigned int lds_addr = (unsigned int)(uintptr_t)lp;
        unsigned long long gaddr = (unsigned long long)(uintptr_t)gp;
        asm volatile("global_load_async_to_lds_b32 %0, %1, off"
                     :: "v"(lds_addr), "v"(gaddr)
                     : "memory");
      } else {
        *lp = 0;
      }
    }
  }
  asm volatile("s_wait_asynccnt 0" ::: "memory");
  __syncthreads();

  const int wave = t >> 5;
  const int lane = t & 31;
  const int M    = lane & 15;
  const int half = lane >> 4;
  const int m0   = wave << 4;
  const int* wdw = (const int*)wpk;

  v8i acc[4];
  acc[0] = (v8i){}; acc[1] = (v8i){}; acc[2] = (v8i){}; acc[3] = (v8i){};

  #pragma unroll
  for (int tap = 0; tap < 9; ++tap) {
    const int kh = tap / 3, kw = tap % 3;
    #pragma unroll
    for (int ch = 0; ch < 2; ++ch) {
      const int* ap = wdw + (((m0 + M) * 9 + tap) * 32) + ch * 16 + half * 4;
      v8i A;
      #pragma unroll
      for (int g = 0; g < 4; ++g) {
        v2i a2 = *(const v2i*)(ap + (g & 1) * 2 + (g >> 1) * 8);
        A[2 * g]     = a2.x;
        A[2 * g + 1] = a2.y;
      }
      #pragma unroll
      for (int nt = 0; nt < 4; ++nt) {
        const int* bp =
            &acts[(kh * 66 + nt * 16 + M + kw) * 36 + ch * 16 + half * 4];
        v4i blo = *(const v4i*)bp;
        v4i bhi = *(const v4i*)(bp + 8);
        v8i Bm;
        Bm[0] = blo.x; Bm[1] = blo.y; Bm[2] = blo.z; Bm[3] = blo.w;
        Bm[4] = bhi.x; Bm[5] = bhi.y; Bm[6] = bhi.z; Bm[7] = bhi.w;
        acc[nt] = __builtin_amdgcn_wmma_i32_16x16x64_iu8(
             true, A,  true, Bm, acc[nt],
             false,  false);
      }
    }
  }

  const float xs = x_scale[0];
  float sc[8], bi[8];
  #pragma unroll
  for (int r = 0; r < 8; ++r) {
    int co = m0 + r + half * 8;
    sc[r] = xs * w_scale[co];
    bi[r] = bias[co];
  }
  #pragma unroll
  for (int nt = 0; nt < 4; ++nt) {
    #pragma unroll
    for (int r = 0; r < 8; ++r) {
      float o = (float)acc[nt][r] * sc[r] + bi[r];
      int co  = m0 + r + half * 8;
      out[(((size_t)b * COUT + co) * HH + h) * WW + w0 + nt * 16 + M] = o;
    }
  }
}

extern "C" void kernel_launch(void* const* d_in, const int* in_sizes, int n_in,
                              void* d_out, int out_size, void* d_ws, size_t ws_size,
                              hipStream_t stream) {
  const float* x        = (const float*)d_in[0];
  const int*   w8       = (const int*)d_in[1];
  const float* w_scale  = (const float*)d_in[2];
  const float* bias     = (const float*)d_in[3];
  const float* x_scale  = (const float*)d_in[4];
  float*       out      = (float*)d_out;

  uint8_t* xq  = (uint8_t*)d_ws;
  uint8_t* wpk = xq + XQ_BYTES;

  quant_act_kernel<<<BB * HH * (WW / 64), 256, 0, stream>>>(x, x_scale, xq);
  pack_w_kernel<<<(COUT * 9 * 32 + 255) / 256, 256, 0, stream>>>(w8, wpk);
  conv_wmma_kernel<<<BB * HH * (WW / 64), 256, 0, stream>>>(
      xq, wpk, w_scale, bias, x_scale, out);

  (void)in_sizes; (void)n_in; (void)out_size; (void)ws_size;
}
